// DTISite_20761871909531
// MI455X (gfx1250) — matrix operand order repaired and checked by reading, not compiled, not run on hardware
//
#include <hip/hip_runtime.h>
#include <hip/hip_bf16.h>

typedef _Float16 half_t;
typedef __attribute__((ext_vector_type(16))) _Float16 v16h;
typedef __attribute__((ext_vector_type(8)))  float    v8f;

#define HID   128
#define BSZ   32
#define NPN   12800
#define NDG   1280
#define EPN   256000
#define EDN   5120
#define NHEAD 2
#define DKH   64
#define SCALE_H 0.08838834764831845f   /* 1/sqrt(128) */

// ---------------------------------------------------------------- helpers
__device__ __forceinline__ float actf(float x, int act) {
  if (act == 1) return x > 0.f ? x : 0.1f * x;     // leaky_relu(0.1)
  if (act == 2) return x > 0.f ? x : 0.f;          // relu
  if (act == 3) return 1.f / (1.f + __expf(-x));   // sigmoid
  return x;
}
__device__ __forceinline__ unsigned f2ord(float f) {
  unsigned u = __float_as_uint(f);
  return (u & 0x80000000u) ? ~u : (u | 0x80000000u);
}
__device__ __forceinline__ float ord2f(unsigned u) {
  return (u & 0x80000000u) ? __uint_as_float(u & 0x7fffffffu) : __uint_as_float(~u);
}

// ---------------------------------------------------------------- fp32->fp16 weight transpose: WT[n*K+k] = W[k*N+n]
__global__ void wt_conv_kernel(const float* __restrict__ W, half_t* __restrict__ WT, int K, int N) {
  int idx = blockIdx.x * blockDim.x + threadIdx.x;
  if (idx >= K * N) return;
  int k = idx / N, n = idx % N;
  WT[(size_t)n * K + k] = (half_t)W[idx];
}

// ---------------------------------------------------------------- WMMA GEMM, 16x64 per wave (4 accumulators):
// C = act(A[M,K] @ W + bias) (+ add).  WT fp16 [N,K].  M%16==0, N%64==0, K arbitrary.
__global__ __launch_bounds__(256) void gemm_kernel(
    const float* __restrict__ A, const half_t* __restrict__ WT,
    const float* __restrict__ bias, const float* __restrict__ add,
    float* __restrict__ C, int M, int N, int K, int act)
{
  int gw = (blockIdx.x * blockDim.x + threadIdx.x) >> 5;
  int lane = threadIdx.x & 31;
  int ntile = N >> 6;                       // 64-wide column tiles
  int total = (M >> 4) * ntile;
  if (gw >= total) return;
  int tm = gw / ntile, tn = gw % ntile;
  int l15 = lane & 15, hi = lane >> 4;
  const float*  Arow = A  + (size_t)(tm * 16 + l15) * K;
  const half_t* B0   = WT + (size_t)(tn * 64 + l15) * K;
  v8f acc[4] = {};
  int kk = 0;
  for (; kk + 32 <= K; kk += 32) {
    v16h a;
    int ka = kk + hi * 16;
#pragma unroll
    for (int e = 0; e < 8; ++e)  a[e]     = (half_t)Arow[ka + e];
#pragma unroll
    for (int e = 0; e < 8; ++e)  a[8 + e] = (half_t)Arow[ka + 8 + e];
    int kb = kk + hi * 16;
#pragma unroll
    for (int t = 0; t < 4; ++t) {
      const half_t* Bt = B0 + (size_t)(16 * t) * K;
      v16h b;
#pragma unroll
      for (int e = 0; e < 16; ++e) b[e] = Bt[kb + e];
      acc[t] = __builtin_amdgcn_wmma_f32_16x16x32_f16(false, a, false, b, (short)0, acc[t], false, false);
    }
  }
  if (kk < K) {   // zero-padded tail
    v16h a;
    int ka = kk + hi * 16, kb = kk + hi * 16;
#pragma unroll
    for (int e = 0; e < 8; ++e)  { int k = ka + e;      a[e]     = (k < K) ? (half_t)Arow[k] : (half_t)0.f; }
#pragma unroll
    for (int e = 0; e < 8; ++e)  { int k = ka + 8 + e; a[8 + e] = (k < K) ? (half_t)Arow[k] : (half_t)0.f; }
#pragma unroll
    for (int t = 0; t < 4; ++t) {
      const half_t* Bt = B0 + (size_t)(16 * t) * K;
      v16h b;
#pragma unroll
      for (int e = 0; e < 16; ++e) { int k = kb + e; b[e] = (k < K) ? Bt[k] : (half_t)0.f; }
      acc[t] = __builtin_amdgcn_wmma_f32_16x16x32_f16(false, a, false, b, (short)0, acc[t], false, false);
    }
  }
#pragma unroll
  for (int t = 0; t < 4; ++t) {
    int col = tn * 64 + 16 * t + l15;
    float bv = bias ? bias[col] : 0.f;
#pragma unroll
    for (int v = 0; v < 8; ++v) {
      int r = tm * 16 + v + hi * 8;
      float val = actf(acc[t][v] + bv, act);
      if (add) val += add[(size_t)r * N + col];
      C[(size_t)r * N + col] = val;
    }
  }
}

// ---------------------------------------------------------------- fused dual-GEMM row-dot:
// out[row] += scale * sum_n ( (A@Wq+bq)[row,n] * (A@Wk+bk)[row,n] )   (K%32==0)
__global__ __launch_bounds__(256) void gemm_rowdot_kernel(
    const float* __restrict__ A,
    const half_t* __restrict__ WqT, const float* __restrict__ bq,
    const half_t* __restrict__ WkT, const float* __restrict__ bk,
    float* __restrict__ out, int M, int N, int K, float scale)
{
  int gw = (blockIdx.x * blockDim.x + threadIdx.x) >> 5;
  int lane = threadIdx.x & 31;
  int ntile = N >> 4;
  int total = (M >> 4) * ntile;
  if (gw >= total) return;
  int tm = gw / ntile, tn = gw % ntile;
  int l15 = lane & 15, hi = lane >> 4;
  const float*  Arow = A   + (size_t)(tm * 16 + l15) * K;
  const half_t* Bq   = WqT + (size_t)(tn * 16 + l15) * K;
  const half_t* Bk   = WkT + (size_t)(tn * 16 + l15) * K;
  v8f accq = {}, acck = {};
  for (int kk = 0; kk + 32 <= K; kk += 32) {
    v16h a, b0, b1;
    int ka = kk + hi * 16;
#pragma unroll
    for (int e = 0; e < 8; ++e)  a[e]     = (half_t)Arow[ka + e];
#pragma unroll
    for (int e = 0; e < 8; ++e)  a[8 + e] = (half_t)Arow[ka + 8 + e];
    int kb = kk + hi * 16;
#pragma unroll
    for (int e = 0; e < 16; ++e) { b0[e] = Bq[kb + e]; b1[e] = Bk[kb + e]; }
    accq = __builtin_amdgcn_wmma_f32_16x16x32_f16(false, a, false, b0, (short)0, accq, false, false);
    acck = __builtin_amdgcn_wmma_f32_16x16x32_f16(false, a, false, b1, (short)0, acck, false, false);
  }
  int col = tn * 16 + l15;
  float bqc = bq[col], bkc = bk[col];
#pragma unroll
  for (int v = 0; v < 8; ++v) {
    float p = (accq[v] + bqc) * (acck[v] + bkc);
    p += __shfl_xor(p, 1, 32);
    p += __shfl_xor(p, 2, 32);
    p += __shfl_xor(p, 4, 32);
    p += __shfl_xor(p, 8, 32);
    if (l15 == 0) atomicAdd(&out[tm * 16 + v + hi * 8], p * scale);
  }
}

// ---------------------------------------------------------------- fused message GEMM + scatter, 16x64 per wave:
// per edge e (row), col k:  atomicAdd(aggr[dst[e]][k], alpha[e]*(Vn[src[e]][k] + (eah@We+be)[e][k]))
__global__ __launch_bounds__(256) void msg_scatter_kernel(
    const float* __restrict__ eah, const half_t* __restrict__ WeT, const float* __restrict__ be,
    const float* __restrict__ alpha, const float* __restrict__ Vn,
    const int* __restrict__ src, const int* __restrict__ dst,
    float* __restrict__ aggr, int E, int K)
{
  int gw = (blockIdx.x * blockDim.x + threadIdx.x) >> 5;
  int lane = threadIdx.x & 31;
  int ntile = HID >> 6;                     // 64-wide tiles over HID=128
  int total = (E >> 4) * ntile;
  if (gw >= total) return;
  int tm = gw / ntile, tn = gw % ntile;
  int l15 = lane & 15, hi = lane >> 4;
  const float*  Arow = eah + (size_t)(tm * 16 + l15) * K;
  const half_t* B0   = WeT + (size_t)(tn * 64 + l15) * K;
  v8f acc[4] = {};
  for (int kk = 0; kk + 32 <= K; kk += 32) {
    v16h a;
    int ka = kk + hi * 16;
#pragma unroll
    for (int e = 0; e < 8; ++e)  a[e]     = (half_t)Arow[ka + e];
#pragma unroll
    for (int e = 0; e < 8; ++e)  a[8 + e] = (half_t)Arow[ka + 8 + e];
    int kb = kk + hi * 16;
#pragma unroll
    for (int t = 0; t < 4; ++t) {
      const half_t* Bt = B0 + (size_t)(16 * t) * K;
      v16h b;
#pragma unroll
      for (int e = 0; e < 16; ++e) b[e] = Bt[kb + e];
      acc[t] = __builtin_amdgcn_wmma_f32_16x16x32_f16(false, a, false, b, (short)0, acc[t], false, false);
    }
  }
#pragma unroll
  for (int v = 0; v < 8; ++v) {
    int e = tm * 16 + v + hi * 8;
    float al = alpha[e];
    int s = src[e], d = dst[e];
#pragma unroll
    for (int t = 0; t < 4; ++t) {
      int col = tn * 64 + 16 * t + l15;
      float val = al * (acc[t][v] + be[col] + Vn[(size_t)s * HID + col]);
      atomicAdd(&aggr[(size_t)d * HID + col], val);
    }
  }
}

// ---------------------------------------------------------------- per-edge 128-dim dot (wave per edge)
__global__ void edge_dot_kernel(const float* __restrict__ Q, const float* __restrict__ Kn,
                                const int* __restrict__ src, const int* __restrict__ dst,
                                float* __restrict__ out, int E, float scale)
{
  int gid = blockIdx.x * blockDim.x + threadIdx.x;
  int e = gid >> 5, lane = gid & 31;
  if (e >= E) return;
  const float* q = Q  + (size_t)src[e] * HID;
  const float* k = Kn + (size_t)dst[e] * HID;
  float acc = 0.f;
#pragma unroll
  for (int i = 0; i < 4; ++i) acc += q[lane + 32 * i] * k[lane + 32 * i];
  acc += __shfl_xor(acc, 16, 32);
  acc += __shfl_xor(acc, 8, 32);
  acc += __shfl_xor(acc, 4, 32);
  acc += __shfl_xor(acc, 2, 32);
  acc += __shfl_xor(acc, 1, 32);
  if (lane == 0) out[e] = acc * scale;
}

// ---------------------------------------------------------------- segment softmax pieces
__global__ void seg_max_kernel(const float* __restrict__ a, const int* __restrict__ src,
                               const int* __restrict__ batch, unsigned* __restrict__ m, int E) {
  int e = blockIdx.x * blockDim.x + threadIdx.x;
  if (e >= E) return;
  atomicMax(&m[batch[src[e]]], f2ord(a[e]));
}
__global__ void seg_sum_kernel(const float* __restrict__ a, const int* __restrict__ src,
                               const int* __restrict__ batch, const unsigned* __restrict__ m,
                               float* __restrict__ s, int E) {
  int e = blockIdx.x * blockDim.x + threadIdx.x;
  if (e >= E) return;
  int g = batch[src[e]];
  atomicAdd(&s[g], __expf(a[e] - ord2f(m[g])));
}
__global__ void alpha_kernel(const float* __restrict__ av, const float* __restrict__ ae,
                             const unsigned* __restrict__ mv, const float* __restrict__ sv,
                             const unsigned* __restrict__ me, const float* __restrict__ se,
                             const int* __restrict__ src, const int* __restrict__ batch,
                             float* __restrict__ alpha, int E) {
  int e = blockIdx.x * blockDim.x + threadIdx.x;
  if (e >= E) return;
  int g = batch[src[e]];
  float p1 = __expf(av[e] - ord2f(mv[g])) / sv[g];
  float p2 = __expf(ae[e] - ord2f(me[g])) / se[g];
  alpha[e] = 0.5f * (p1 + p2);
}

// ---------------------------------------------------------------- elementwise utilities
__global__ void fill_f_kernel(float* p, int n, float v) {
  int i = blockIdx.x * blockDim.x + threadIdx.x; if (i < n) p[i] = v;
}
__global__ void fill_u_kernel(unsigned* p, int n, unsigned v) {
  int i = blockIdx.x * blockDim.x + threadIdx.x; if (i < n) p[i] = v;
}
__global__ void copy_kernel(const float* s, float* d, int n) {
  int i = blockIdx.x * blockDim.x + threadIdx.x; if (i < n) d[i] = s[i];
}
__global__ void add_kernel(float* d, const float* s, int n) {   // d += s
  int i = blockIdx.x * blockDim.x + threadIdx.x; if (i < n) d[i] += s[i];
}
__global__ void relu_add_kernel(const float* o, const float* a, float* pf, int n) { // pf = relu(o+a)+pf
  int i = blockIdx.x * blockDim.x + threadIdx.x; if (i >= n) return;
  float v = o[i] + a[i];
  pf[i] = (v > 0.f ? v : 0.f) + pf[i];
}
__global__ void gate_kernel(const float* g, const float* hc, const float* hs, float* out, int n) {
  int i = blockIdx.x * blockDim.x + threadIdx.x; if (i >= n) return;
  float gg = g[i];
  out[i] = gg * hc[i] + (1.f - gg) * hs[i];
}
__global__ void concat_kernel(const float* a, int wa, const float* b, int wb, float* out, int rows) {
  int w = wa + wb;
  long long idx = (long long)blockIdx.x * blockDim.x + threadIdx.x;
  if (idx >= (long long)rows * w) return;
  int r = (int)(idx / w), c = (int)(idx % w);
  out[idx] = (c < wa) ? a[(size_t)r * wa + c] : b[(size_t)r * wb + (c - wa)];
}
__global__ void gather_concat_kernel(const float* x, const float* ea, const int* src, float* out, int E) {
  int idx = blockIdx.x * blockDim.x + threadIdx.x;
  if (idx >= E * 134) return;
  int e = idx / 134, c = idx % 134;
  out[idx] = (c < HID) ? x[(size_t)src[e] * HID + c] : ea[(size_t)e * 6 + (c - HID)];
}
__global__ void scatter_add_kernel(const float* m, const int* dst, float* out, int E, int H) {
  int idx = blockIdx.x * blockDim.x + threadIdx.x;
  if (idx >= E * H) return;
  int e = idx / H, k = idx % H;
  atomicAdd(&out[(size_t)dst[e] * H + k], m[idx]);
}

// ---------------------------------------------------------------- block-diagonal cross-attention (contiguous segments)
__global__ void attn_kernel(const float* __restrict__ Q, const float* __restrict__ Kt,
                            const float* __restrict__ V, float* __restrict__ ctx,
                            int qpg, int kpg)
{
  __shared__ float Ks[64][DKH + 1];
  __shared__ float Vs[64][DKH + 1];
  int qchunks = (qpg + 63) >> 6;
  int bid = blockIdx.x;
  int qc = bid % qchunks;
  int r  = bid / qchunks;
  int h  = r % NHEAD;
  int g  = r / NHEAD;
  int ql = qc * 64 + (int)threadIdx.x;
  bool qv = ql < qpg;
  int qrow = g * qpg + ql;
  float qreg[DKH];
  if (qv) {
#pragma unroll
    for (int d = 0; d < DKH; ++d) qreg[d] = Q[(size_t)qrow * HID + h * DKH + d];
  }
  float m = -3.0e38f, s = 0.f, acc[DKH];
#pragma unroll
  for (int d = 0; d < DKH; ++d) acc[d] = 0.f;
  for (int kc = 0; kc < kpg; kc += 64) {
    int cn = kpg - kc; if (cn > 64) cn = 64;
    __syncthreads();
    if ((int)threadIdx.x < cn) {
      int kr = g * kpg + kc + (int)threadIdx.x;
#pragma unroll
      for (int d = 0; d < DKH; ++d) {
        Ks[threadIdx.x][d] = Kt[(size_t)kr * HID + h * DKH + d];
        Vs[threadIdx.x][d] = V [(size_t)kr * HID + h * DKH + d];
      }
    }
    __syncthreads();
    if (qv) {
      for (int j = 0; j < cn; ++j) {
        float sc = 0.f;
#pragma unroll
        for (int d = 0; d < DKH; ++d) sc += qreg[d] * Ks[j][d];
        sc *= 0.125f;                        // 1/sqrt(64)
        float mn = fmaxf(m, sc);
        float corr = __expf(m - mn), w = __expf(sc - mn);
        s = s * corr + w;
#pragma unroll
        for (int d = 0; d < DKH; ++d) acc[d] = acc[d] * corr + w * Vs[j][d];
        m = mn;
      }
    }
  }
  if (qv) {
    float inv = 1.f / s;
#pragma unroll
    for (int d = 0; d < DKH; ++d) ctx[(size_t)qrow * HID + h * DKH + d] = acc[d] * inv;
  }
}

// ---------------------------------------------------------------- LayerNorm over last dim (H=128), one row per block
__global__ void ln_kernel(const float* __restrict__ x, const float* __restrict__ g,
                          const float* __restrict__ b, float* __restrict__ y)
{
  __shared__ float red[HID];
  int row = blockIdx.x, t = threadIdx.x;
  float v = x[(size_t)row * HID + t];
  red[t] = v; __syncthreads();
  for (int o = 64; o > 0; o >>= 1) { if (t < o) red[t] += red[t + o]; __syncthreads(); }
  float mu = red[0] * (1.f / HID);
  __syncthreads();
  float dv = (v - mu) * (v - mu);
  red[t] = dv; __syncthreads();
  for (int o = 64; o > 0; o >>= 1) { if (t < o) red[t] += red[t + o]; __syncthreads(); }
  float var = red[0] * (1.f / HID);
  y[(size_t)row * HID + t] = (v - mu) * rsqrtf(var + 1e-5f) * g[t] + b[t];
}

// ---------------------------------------------------------------- segment mean+max pool (contiguous), z row = [mean|max] at zoff
__global__ void pool_kernel(const float* __restrict__ h, int rows, float* __restrict__ z, int zoff) {
  int g = blockIdx.x, k = threadIdx.x;
  const float* base = h + (size_t)g * rows * HID;
  float s = 0.f, mx = -3.0e38f;
  for (int r = 0; r < rows; ++r) { float v = base[(size_t)r * HID + k]; s += v; mx = fmaxf(mx, v); }
  z[(size_t)g * 512 + zoff + k]       = s / (float)rows;
  z[(size_t)g * 512 + zoff + HID + k] = mx;
}

// ---------------------------------------------------------------- small VALU matmul (surf path, classifier)
__global__ void smallmm_kernel(const float* __restrict__ A, const float* __restrict__ W,
                               const float* __restrict__ bias, float* __restrict__ C,
                               int M, int N, int K, int act)
{
  int idx = blockIdx.x * blockDim.x + threadIdx.x;
  if (idx >= M * N) return;
  int m = idx / N, n = idx % N;
  float acc = bias ? bias[n] : 0.f;
  for (int k = 0; k < K; ++k) acc += A[(size_t)m * K + k] * W[(size_t)k * N + n];
  C[idx] = actf(acc, act);
}

// ================================================================ host
extern "C" void kernel_launch(void* const* d_in, const int* in_sizes, int n_in,
                              void* d_out, int out_size, void* d_ws, size_t ws_size,
                              hipStream_t stream)
{
  (void)in_sizes; (void)n_in; (void)out_size; (void)ws_size;
  auto F = [&](int i) -> const float* { return (const float*)d_in[i]; };
  auto I = [&](int i) -> const int*   { return (const int*)d_in[i]; };

  // data inputs (setup_inputs order)
  const float* protein_x = F(0);
  const float* surface_x = F(1);
  const int*   pro_src   = I(2);
  const int*   pro_dst   = I(2) + EPN;
  const float* pro_ea    = F(3);
  const int*   pro_batch = I(4);
  const float* drug_x    = F(5);
  const int*   dr_src    = I(6);
  const int*   dr_dst    = I(6) + EDN;
  const float* drug_ea   = F(7);
  const int*   dr_batch  = I(8);  (void)dr_batch;
  // params flattened jax-tree style (sorted dict keys) starting at index 9 (see analysis)

  char* wp = (char*)d_ws;
  auto af = [&](size_t n) -> float*  { float*  p = (float*) wp; wp += ((n * 4 + 255) & ~(size_t)255); return p; };
  auto ah = [&](size_t n) -> half_t* { half_t* p = (half_t*)wp; wp += ((n * 2 + 255) & ~(size_t)255); return p; };
  auto GR = [&](long long n, int b) -> int { return (int)((n + b - 1) / b); };

  auto GEMM = [&](const float* A, const half_t* Wt, const float* bias, const float* add,
                  float* C, int M, int N, int K, int act) {
    long long waves = (long long)(M / 16) * (N / 64);   // 16x64 tiles per wave
    gemm_kernel<<<GR(waves, 8), 256, 0, stream>>>(A, Wt, bias, add, C, M, N, K, act);
  };
  auto MKWT = [&](int widx, int K, int N) -> half_t* {
    half_t* t = ah((size_t)K * N);
    wt_conv_kernel<<<GR((long long)K * N, 256), 256, 0, stream>>>(F(widx), t, K, N);
    return t;
  };

  // ---- convert all WMMA weights once (persistent)
  half_t* wt_pe1  = MKWT(106, 1024, HID);
  half_t* wt_pe2  = MKWT(108, HID, HID);
  half_t* wt_cat1 = MKWT(10, 160, HID);
  half_t* wt_cat2 = MKWT(12, HID, HID);
  half_t* wt_demb = MKWT(52, 82, HID);
  half_t* wt_dg_u1[2] = { MKWT(54, 256, HID), MKWT(58, 256, HID) };
  half_t* wt_dg_u2[2] = { MKWT(56, 134, HID), MKWT(60, 134, HID) };
  struct PG { half_t *le,*out,*we,*wke,*wkn,*wqe,*wqn,*wv;
              const float *le_b,*out_b,*we_b,*wke_b,*wkn_b,*wqe_b,*wqn_b,*wv_b; };
  PG pg[2];
  int pgbase[2] = {73, 89};
  for (int L = 0; L < 2; ++L) {
    int o = pgbase[L];
    pg[L].le_b  = F(o + 0);  pg[L].le  = MKWT(o + 1, 8, HID);
    pg[L].out_b = F(o + 2);  pg[L].out = MKWT(o + 3, HID, HID);
    pg[L].we_b  = F(o + 4);  pg[L].we  = MKWT(o + 5, HID, HID);
    pg[L].wke_b = F(o + 6);  pg[L].wke = MKWT(o + 7, HID, HID);
    pg[L].wkn_b = F(o + 8);  pg[L].wkn = MKWT(o + 9, HID, HID);
    pg[L].wqe_b = F(o + 10); pg[L].wqe = MKWT(o + 11, HID, HID);
    pg[L].wqn_b = F(o + 12); pg[L].wqn = MKWT(o + 13, HID, HID);
    pg[L].wv_b  = F(o + 14); pg[L].wv  = MKWT(o + 15, HID, HID);
  }
  half_t* cp_wk = MKWT(36, HID, HID); half_t* cp_wo = MKWT(38, HID, HID);
  half_t* cp_wq = MKWT(40, HID, HID); half_t* cp_wv = MKWT(42, HID, HID);
  half_t* cd_wk = MKWT(16, HID, HID); half_t* cd_wo = MKWT(18, HID, HID);
  half_t* cd_wq = MKWT(20, HID, HID); half_t* cd_wv = MKWT(22, HID, HID);
  half_t* wt_gp = MKWT(64, 256, HID); half_t* wt_gd = MKWT(62, 256, HID);
  half_t* wt_pm1 = MKWT(68, HID, 256); half_t* wt_pm2 = MKWT(70, 256, HID); half_t* wt_pres = MKWT(72, HID, HID);
  half_t* wt_dm1 = MKWT(46, HID, 256); half_t* wt_dm2 = MKWT(48, 256, HID); half_t* wt_dres = MKWT(50, HID, HID);

  // ---- persistent activations
  float* surf   = af((size_t)NPN * 32);
  float* pf     = af((size_t)NPN * HID);
  float* p_init = af((size_t)NPN * HID);
  float* pfc    = af((size_t)NPN * HID);
  float* df     = af((size_t)NDG * HID);
  float* d_init = af((size_t)NDG * HID);
  float* Hp     = af((size_t)NPN * HID);
  float* Hd     = af((size_t)NDG * HID);
  float* Hpc    = af((size_t)NPN * HID);
  float* Hdc    = af((size_t)NDG * HID);
  float* HpO    = af((size_t)NPN * HID);
  float* HdO    = af((size_t)NDG * HID);
  float* z      = af((size_t)BSZ * 512);
  char* scratch = wp;

  // ================= phase A: surface + protein embedding
  wp = scratch;
  {
    float* tS = af((size_t)NPN * 32);
    float* hA = af((size_t)NPN * HID);
    smallmm_kernel<<<GR((long long)NPN * 32, 256), 256, 0, stream>>>(surface_x, F(110), F(109), tS, NPN, 32, 10, 1);
    smallmm_kernel<<<GR((long long)NPN * 32, 256), 256, 0, stream>>>(tS, F(112), F(111), surf, NPN, 32, 32, 0);
    GEMM(protein_x, wt_pe1, F(105), nullptr, hA, NPN, HID, 1024, 1);
    GEMM(hA, wt_pe2, F(107), nullptr, pf, NPN, HID, HID, 0);
    copy_kernel<<<GR((long long)NPN * HID, 256), 256, 0, stream>>>(pf, p_init, NPN * HID);
  }

  // ================= phase B: protein GNN (2 layers)
  for (int L = 0; L < 2; ++L) {
    wp = scratch;
    float* eah  = af((size_t)EPN * HID);
    float* Qn   = af((size_t)NPN * HID);
    float* Kn   = af((size_t)NPN * HID);
    float* Vn   = af((size_t)NPN * HID);
    float* OutL = af((size_t)NPN * HID);
    float* aggr = af((size_t)NPN * HID);
    float* a_v  = af(EPN);
    float* a_e  = af(EPN);
    float* alp  = af(EPN);
    unsigned* mv = (unsigned*)af(BSZ); float* sv = af(BSZ);
    unsigned* me = (unsigned*)af(BSZ); float* se = af(BSZ);

    GEMM(pro_ea, pg[L].le, pg[L].le_b, nullptr, eah, EPN, HID, 8, 2);        // relu(lin_edge)
    GEMM(pf, pg[L].wqn, pg[L].wqn_b, nullptr, Qn,   NPN, HID, HID, 0);       // node-level Q/K/V/out
    GEMM(pf, pg[L].wkn, pg[L].wkn_b, nullptr, Kn,   NPN, HID, HID, 0);
    GEMM(pf, pg[L].wv,  pg[L].wv_b,  nullptr, Vn,   NPN, HID, HID, 0);
    GEMM(pf, pg[L].out, pg[L].out_b, nullptr, OutL, NPN, HID, HID, 0);

    fill_f_kernel<<<GR(EPN, 256), 256, 0, stream>>>(a_e, EPN, 0.f);
    {
      long long waves = (long long)(EPN / 16) * (HID / 16);
      gemm_rowdot_kernel<<<GR(waves, 8), 256, 0, stream>>>(eah, pg[L].wqe, pg[L].wqe_b,
                                                           pg[L].wke, pg[L].wke_b,
                                                           a_e, EPN, HID, HID, SCALE_H);
    }
    edge_dot_kernel<<<GR((long long)EPN * 32, 256), 256, 0, stream>>>(Qn, Kn, pro_src, pro_dst, a_v, EPN, SCALE_H);

    fill_u_kernel<<<1, 64, 0, stream>>>(mv, BSZ, 0x007FFFFFu);   // ord(-inf)
    fill_u_kernel<<<1, 64, 0, stream>>>(me, BSZ, 0x007FFFFFu);
    fill_f_kernel<<<1, 64, 0, stream>>>(sv, BSZ, 0.f);
    fill_f_kernel<<<1, 64, 0, stream>>>(se, BSZ, 0.f);
    seg_max_kernel<<<GR(EPN, 256), 256, 0, stream>>>(a_v, pro_src, pro_batch, mv, EPN);
    seg_max_kernel<<<GR(EPN, 256), 256, 0, stream>>>(a_e, pro_src, pro_batch, me, EPN);
    seg_sum_kernel<<<GR(EPN, 256), 256, 0, stream>>>(a_v, pro_src, pro_batch, mv, sv, EPN);
    seg_sum_kernel<<<GR(EPN, 256), 256, 0, stream>>>(a_e, pro_src, pro_batch, me, se, EPN);
    alpha_kernel<<<GR(EPN, 256), 256, 0, stream>>>(a_v, a_e, mv, sv, me, se, pro_src, pro_batch, alp, EPN);

    fill_f_kernel<<<GR((long long)NPN * HID, 256), 256, 0, stream>>>(aggr, NPN * HID, 0.f);
    {
      long long waves = (long long)(EPN / 16) * (HID / 64);
      msg_scatter_kernel<<<GR(waves, 8), 256, 0, stream>>>(eah, pg[L].we, pg[L].we_b, alp, Vn,
                                                           pro_src, pro_dst, aggr, EPN, HID);
    }
    relu_add_kernel<<<GR((long long)NPN * HID, 256), 256, 0, stream>>>(OutL, aggr, pf, NPN * HID);
  }
  add_kernel<<<GR((long long)NPN * HID, 256), 256, 0, stream>>>(pf, p_init, NPN * HID);

  // ================= phase C: concat surface + cat MLP
  wp = scratch;
  {
    float* catb = af((size_t)NPN * 160);
    float* hC   = af((size_t)NPN * HID);
    concat_kernel<<<GR((long long)NPN * 160, 256), 256, 0, stream>>>(pf, HID, surf, 32, catb, NPN);
    GEMM(catb, wt_cat1, F(9), nullptr, hC, NPN, HID, 160, 1);
    GEMM(hC, wt_cat2, F(11), nullptr, pfc, NPN, HID, HID, 0);
  }

  // ================= phase D: drug embedding + GNN (2 layers)
  GEMM(drug_x, wt_demb, F(51), nullptr, df, NDG, HID, 82, 1);
  copy_kernel<<<GR((long long)NDG * HID, 256), 256, 0, stream>>>(df, d_init, NDG * HID);
  for (int L = 0; L < 2; ++L) {
    wp = scratch;
    float* dtmp = af((size_t)EDN * 134);
    float* mb   = af((size_t)EDN * HID);
    float* msg  = af((size_t)NDG * HID);
    float* dc   = af((size_t)NDG * 256);
    gather_concat_kernel<<<GR((long long)EDN * 134, 256), 256, 0, stream>>>(df, drug_ea, dr_src, dtmp, EDN);
    GEMM(dtmp, wt_dg_u2[L], F(L ? 59 : 55), nullptr, mb, EDN, HID, 134, 1);
    fill_f_kernel<<<GR((long long)NDG * HID, 256), 256, 0, stream>>>(msg, NDG * HID, 0.f);
    scatter_add_kernel<<<GR((long long)EDN * HID, 256), 256, 0, stream>>>(mb, dr_dst, msg, EDN, HID);
    concat_kernel<<<GR((long long)NDG * 256, 256), 256, 0, stream>>>(df, HID, msg, HID, dc, NDG);
    GEMM(dc, wt_dg_u1[L], F(L ? 57 : 53), df, df, NDG, HID, 256, 0);  // += residual df
  }
  add_kernel<<<GR((long long)NDG * HID, 256), 256, 0, stream>>>(df, d_init, NDG * HID);

  // ================= phase E: cross-attention (block-diagonal per graph)
  wp = scratch;
  {
    float* Qp = af((size_t)NPN * HID); float* Kd = af((size_t)NDG * HID); float* Vd = af((size_t)NDG * HID);
    float* ctxP = af((size_t)NPN * HID); float* preP = af((size_t)NPN * HID);
    float* Qd = af((size_t)NDG * HID); float* Kp = af((size_t)NPN * HID); float* Vp = af((size_t)NPN * HID);
    float* ctxD = af((size_t)NDG * HID); float* preD = af((size_t)NDG * HID);

    GEMM(pfc, cp_wq, F(39), nullptr, Qp, NPN, HID, HID, 0);
    GEMM(df,  cp_wk, F(35), nullptr, Kd, NDG, HID, HID, 0);
    GEMM(df,  cp_wv, F(41), nullptr, Vd, NDG, HID, HID, 0);
    { int qpg = NPN / BSZ, kpg = NDG / BSZ, qch = (qpg + 63) / 64;
      attn_kernel<<<BSZ * NHEAD * qch, 64, 0, stream>>>(Qp, Kd, Vd, ctxP, qpg, kpg); }
    GEMM(ctxP, cp_wo, F(37), pfc, preP, NPN, HID, HID, 0);
    ln_kernel<<<NPN, HID, 0, stream>>>(preP, F(34), F(33), Hp);

    GEMM(df,  cd_wq, F(19), nullptr, Qd, NDG, HID, HID, 0);
    GEMM(pfc, cd_wk, F(15), nullptr, Kp, NPN, HID, HID, 0);
    GEMM(pfc, cd_wv, F(21), nullptr, Vp, NPN, HID, HID, 0);
    { int qpg = NDG / BSZ, kpg = NPN / BSZ, qch = (qpg + 63) / 64;
      attn_kernel<<<BSZ * NHEAD * qch, 64, 0, stream>>>(Qd, Kp, Vp, ctxD, qpg, kpg); }
    GEMM(ctxD, cd_wo, F(17), df, preD, NDG, HID, HID, 0);
    ln_kernel<<<NDG, HID, 0, stream>>>(preD, F(14), F(13), Hd);
  }

  // ================= phase F: gated fusion + MLPs
  wp = scratch;
  {
    float* gcp = af((size_t)NPN * 256); float* gP = af((size_t)NPN * HID);
    float* gcd = af((size_t)NDG * 256); float* gD = af((size_t)NDG * HID);
    float* h1p = af((size_t)NPN * 256); float* h2p = af((size_t)NPN * HID); float* lnp = af((size_t)NPN * HID);
    float* h1d = af((size_t)NDG * 256); float* h2d = af((size_t)NDG * HID); float* lnd = af((size_t)NDG * HID);

    concat_kernel<<<GR((long long)NPN * 256, 256), 256, 0, stream>>>(Hp, HID, pfc, HID, gcp, NPN);
    GEMM(gcp, wt_gp, F(63), nullptr, gP, NPN, HID, 256, 3);
    gate_kernel<<<GR((long long)NPN * HID, 256), 256, 0, stream>>>(gP, Hp, pfc, Hpc, NPN * HID);
    concat_kernel<<<GR((long long)NDG * 256, 256), 256, 0, stream>>>(Hd, HID, df, HID, gcd, NDG);
    GEMM(gcd, wt_gd, F(61), nullptr, gD, NDG, HID, 256, 3);
    gate_kernel<<<GR((long long)NDG * HID, 256), 256, 0, stream>>>(gD, Hd, df, Hdc, NDG * HID);

    GEMM(Hpc, wt_pm1, F(67), nullptr, h1p, NPN, 256, HID, 1);
    GEMM(h1p, wt_pm2, F(69), nullptr, h2p, NPN, HID, 256, 0);
    ln_kernel<<<NPN, HID, 0, stream>>>(h2p, F(66), F(65), lnp);
    GEMM(Hpc, wt_pres, F(71), lnp, HpO, NPN, HID, HID, 0);

    GEMM(Hdc, wt_dm1, F(45), nullptr, h1d, NDG, 256, HID, 1);
    GEMM(h1d, wt_dm2, F(47), nullptr, h2d, NDG, HID, 256, 0);
    ln_kernel<<<NDG, HID, 0, stream>>>(h2d, F(44), F(43), lnd);
    GEMM(Hdc, wt_dres, F(49), lnd, HdO, NDG, HID, HID, 0);
  }

  // ================= phase G: pooling + classifier
  pool_kernel<<<BSZ, HID, 0, stream>>>(HpO, NPN / BSZ, z, 0);
  pool_kernel<<<BSZ, HID, 0, stream>>>(HdO, NDG / BSZ, z, 256);
  wp = scratch;
  {
    float* z1  = af((size_t)BSZ * 256);
    float* z2  = af((size_t)BSZ * HID);
    float* z2l = af((size_t)BSZ * HID);
    float* z3  = af((size_t)BSZ * 64);
    smallmm_kernel<<<GR((long long)BSZ * 256, 256), 256, 0, stream>>>(z,  F(24), F(23), z1, BSZ, 256, 512, 1);
    smallmm_kernel<<<GR((long long)BSZ * 128, 256), 256, 0, stream>>>(z1, F(26), F(25), z2, BSZ, 128, 256, 1);
    ln_kernel<<<BSZ, HID, 0, stream>>>(z2, F(32), F(31), z2l);
    smallmm_kernel<<<GR((long long)BSZ * 64, 256), 256, 0, stream>>>(z2l, F(28), F(27), z3, BSZ, 64, 128, 1);
    smallmm_kernel<<<1, 64, 0, stream>>>(z3, F(30), F(29), (float*)d_out, BSZ, 1, 64, 3);
  }
}
